// WavKANClassifier_79585743995016
// MI455X (gfx1250) — compile-verified, matrix operand order checked by reading, not run on hardware
//
#include <hip/hip_runtime.h>
#include <hip/hip_bf16.h>

// ---------------------------------------------------------------------------
// WavKAN classifier: 3x (wavelet "linear" -> silu -> layernorm) -> f16-WMMA GEMM
// B=512, IN=1024, HID=512, NCLS=1000
// Wavelet layers: transcendental/VALU-bound -> LDS tiling + async global->LDS
// DMA double-buffering (ASYNCcnt) + global_prefetch.  Classifier: v_wmma f16
// with f32 accumulate, N padded to 1024 zero rows so EXEC stays all-1s.
// ---------------------------------------------------------------------------

typedef _Float16 v8h  __attribute__((ext_vector_type(8)));
typedef _Float16 v16h __attribute__((ext_vector_type(16)));
typedef float    v8f  __attribute__((ext_vector_type(8)));

#define WB   512
#define WHID 512
#define WIN0 1024
#define WNC  1000
#define WNCP 1024   // NCLS padded to multiple of 64

// Async global->LDS path (gfx1250): guarded so either toolchain compiles.
#if defined(__has_builtin)
#if __has_builtin(__builtin_amdgcn_global_load_async_to_lds_b128) && \
    __has_builtin(__builtin_amdgcn_s_wait_asynccnt)
#define HAVE_ASYNC_LDS 1
#endif
#endif

#ifdef HAVE_ASYNC_LDS
// Builtin signature (from hipcc diagnostic): params are pointers to
// 'int __attribute__((vector_size(16)))' (global src, LDS dst, imm offset, imm cpol)
typedef int async_v4i __attribute__((vector_size(4 * sizeof(int))));
#define GLBP(p) ((__attribute__((address_space(1))) async_v4i*)(p))
#define LDSP(p) ((__attribute__((address_space(3))) async_v4i*)(p))
#endif

// ------------------------- wavelet layer (VALU-bound) ----------------------
// block = 256 threads, tile = 64 batches x 16 outs, IN chunked by 32,
// double-buffered LDS staging.
#define BT 64
#define OT 16
#define NT 32

__global__ __launch_bounds__(256)
void wavelet_kernel(const float* __restrict__ x, const float* __restrict__ t,
                    const float* __restrict__ sc, const float* __restrict__ w,
                    float* __restrict__ out, int IN) {
  __shared__ __align__(16) float xs[2][BT][NT];   // 16 KB, broadcast reads
  __shared__ float ts [2][OT][NT + 1];            // +1 pad: 33 & 64 coprime
  __shared__ float ssh[2][OT][NT + 1];
  __shared__ float wsh[2][OT][NT + 1];

  const int tid  = threadIdx.x;
  const int o0   = blockIdx.x * OT;
  const int b0   = blockIdx.y * BT;
  const int osub = tid & (OT - 1);
  const int bq   = tid >> 4;          // 0..15, owns batches bq*4 .. bq*4+3

  const int xr = tid >> 2;            // cooperative x-tile load: row
  const int xc = (tid & 3) * 8;       // 8 floats (two 16B granules)
  const int pr = tid >> 4;            // cooperative param load: row
  const int pc = (tid & 15) * 2;      // 2 floats

  const float* xg = x  + (size_t)(b0 + xr) * IN + xc;
  const float* tg = t  + (size_t)(o0 + pr) * IN + pc;
  const float* sg = sc + (size_t)(o0 + pr) * IN + pc;
  const float* wg = w  + (size_t)(o0 + pr) * IN + pc;

  float acc0 = 0.f, acc1 = 0.f, acc2 = 0.f, acc3 = 0.f;

  // stage chunk n0 into LDS buffer `bf` (x via async DMA when available)
  auto stage = [&](int bf, int n0) {
#ifdef HAVE_ASYNC_LDS
    __builtin_amdgcn_global_load_async_to_lds_b128(
        GLBP(xg + n0),     LDSP(&xs[bf][xr][xc]),     0, 0);
    __builtin_amdgcn_global_load_async_to_lds_b128(
        GLBP(xg + n0 + 4), LDSP(&xs[bf][xr][xc + 4]), 0, 0);
#else
    const float4* xp = (const float4*)(xg + n0);
    float4 xa = xp[0], xb = xp[1];
    *(float4*)&xs[bf][xr][xc]     = xa;
    *(float4*)&xs[bf][xr][xc + 4] = xb;
#endif
    float2 tv = *(const float2*)(tg + n0);
    float2 sv = *(const float2*)(sg + n0);
    float2 wv = *(const float2*)(wg + n0);
    ts [bf][pr][pc] = tv.x;  ts [bf][pr][pc + 1] = tv.y;
    ssh[bf][pr][pc] = sv.x;  ssh[bf][pr][pc + 1] = sv.y;
    wsh[bf][pr][pc] = wv.x;  wsh[bf][pr][pc + 1] = wv.y;
    // warm L2/WGP$ one chunk ahead of the staging (global_prefetch_b8)
    if (n0 + NT < IN) {
      __builtin_prefetch(xg + n0 + NT, 0, 3);
      __builtin_prefetch(tg + n0 + NT, 0, 3);
      __builtin_prefetch(sg + n0 + NT, 0, 3);
      __builtin_prefetch(wg + n0 + NT, 0, 3);
    }
  };

  stage(0, 0);
  int bf = 0;
#pragma unroll 2
  for (int n0 = 0; n0 < IN; n0 += NT, bf ^= 1) {
#ifdef HAVE_ASYNC_LDS
    __builtin_amdgcn_s_wait_asynccnt(0);  // my DMA into xs[bf] complete
#endif
    __syncthreads();                      // everyone's staging of bf complete;
                                          // also: all reads of bf^1 finished
    if (n0 + NT < IN) stage(bf ^ 1, n0 + NT);

#pragma unroll 8
    for (int nn = 0; nn < NT; ++nn) {
      float tt  = ts[bf][osub][nn];
      float rr  = __builtin_amdgcn_rcpf(ssh[bf][osub][nn]); // v_rcp_f32, /4 amortized
      float ww  = wsh[bf][osub][nn];
      float r2h = -0.5f * (rr * rr);                        // p = d*d*r2h = -s^2/2
      float d0 = xs[bf][bq * 4 + 0][nn] - tt;
      float d1 = xs[bf][bq * 4 + 1][nn] - tt;
      float d2 = xs[bf][bq * 4 + 2][nn] - tt;
      float d3 = xs[bf][bq * 4 + 3][nn] - tt;
      float p0 = d0 * d0 * r2h;
      float p1 = d1 * d1 * r2h;
      float p2 = d2 * d2 * r2h;
      float p3 = d3 * d3 * r2h;
      float u0 = __builtin_fmaf(2.f, p0, 1.f);              // 1 - s^2
      float u1 = __builtin_fmaf(2.f, p1, 1.f);
      float u2 = __builtin_fmaf(2.f, p2, 1.f);
      float u3 = __builtin_fmaf(2.f, p3, 1.f);
      acc0 += ww * u0 * __expf(p0);                         // v_exp_f32
      acc1 += ww * u1 * __expf(p1);
      acc2 += ww * u2 * __expf(p2);
      acc3 += ww * u3 * __expf(p3);
    }
  }

  const int oc = o0 + osub;
  out[(size_t)(b0 + bq * 4 + 0) * WHID + oc] = acc0;
  out[(size_t)(b0 + bq * 4 + 1) * WHID + oc] = acc1;
  out[(size_t)(b0 + bq * 4 + 2) * WHID + oc] = acc2;
  out[(size_t)(b0 + bq * 4 + 3) * WHID + oc] = acc3;
}

// ------------------------- fused SiLU + LayerNorm --------------------------
template <typename OutT>
__global__ __launch_bounds__(256)
void silu_ln_kernel(const float* __restrict__ h, const float* __restrict__ g,
                    const float* __restrict__ bta, OutT* __restrict__ f) {
  const int row = blockIdx.x;
  const int tid = threadIdx.x;
  const float* hr = h + (size_t)row * WHID;

  float v0 = hr[tid];
  float v1 = hr[tid + 256];
  float y0 = v0 / (1.0f + __expf(-v0));
  float y1 = v1 / (1.0f + __expf(-v1));

  float s = y0 + y1;
  float q = y0 * y0 + y1 * y1;
#pragma unroll
  for (int off = 16; off > 0; off >>= 1) {   // wave32 tree
    s += __shfl_down(s, off);
    q += __shfl_down(q, off);
  }
  __shared__ float rs[8], rq[8];
  __shared__ float mu_s, inv_s;
  if ((tid & 31) == 0) { rs[tid >> 5] = s; rq[tid >> 5] = q; }
  __syncthreads();
  if (tid == 0) {
    float S = 0.f, Q = 0.f;
#pragma unroll
    for (int i = 0; i < 8; ++i) { S += rs[i]; Q += rq[i]; }
    float mu  = S * (1.0f / WHID);
    float var = Q * (1.0f / WHID) - mu * mu;
    mu_s  = mu;
    inv_s = rsqrtf(var + 1e-5f);
  }
  __syncthreads();
  float mu = mu_s, inv = inv_s;
  f[(size_t)row * WHID + tid]       = (OutT)((y0 - mu) * inv * g[tid]       + bta[tid]);
  f[(size_t)row * WHID + tid + 256] = (OutT)((y1 - mu) * inv * g[tid + 256] + bta[tid + 256]);
}

// --------------- cls_w f32 -> f16, padded to 1024 rows of zeros ------------
__global__ __launch_bounds__(256)
void convert_w_f16(const float* __restrict__ wsrc, _Float16* __restrict__ wh) {
  int idx = blockIdx.x * 256 + threadIdx.x;     // 0 .. 1024*512-1
  int row = idx >> 9;
  wh[idx] = (row < WNC) ? (_Float16)wsrc[idx] : (_Float16)0.0f;
}

// --------------------- classifier GEMM via v_wmma --------------------------
// D(512x1000) = Fh(512x512) * Wh^T + bias.  One wave -> 16x64 output tile.
// 16-bit A/B layout: lanes 0-15 hold K 0-7 & 16-23, lanes 16-31 K 8-15 & 24-31
// => two contiguous 16B loads per lane per operand per K-step of 32.
__global__ __launch_bounds__(256)
void cls_gemm_wmma(const _Float16* __restrict__ Fh,   // 512 x 512
                   const _Float16* __restrict__ Wh,   // 1024 x 512 (zero-padded)
                   const float* __restrict__ bias,    // 1000
                   float* __restrict__ out) {         // 512 x 1000
  const int lane   = threadIdx.x & 31;
  const int wid    = (blockIdx.x << 3) | (threadIdx.x >> 5);
  const int tileM  = wid & 31;        // 32 tiles of 16 rows
  const int grp    = wid >> 5;        // 16 groups of 64 cols
  const int sel    = lane >> 4;       // half-wave select
  const int mn     = lane & 15;

  const _Float16* aBase  = Fh + (size_t)(tileM * 16 + mn) * WHID + sel * 8;
  const _Float16* bBase0 = Wh + (size_t)(grp * 64 +  0 + mn) * WHID + sel * 8;
  const _Float16* bBase1 = Wh + (size_t)(grp * 64 + 16 + mn) * WHID + sel * 8;
  const _Float16* bBase2 = Wh + (size_t)(grp * 64 + 32 + mn) * WHID + sel * 8;
  const _Float16* bBase3 = Wh + (size_t)(grp * 64 + 48 + mn) * WHID + sel * 8;

  v8f acc[4] = {};

  for (int k = 0; k < WHID; k += 32) {
    v8h al = *(const v8h*)(aBase + k);
    v8h ah = *(const v8h*)(aBase + k + 16);
    v16h a = __builtin_shufflevector(al, ah, 0,1,2,3,4,5,6,7,8,9,10,11,12,13,14,15);

    v8h b0l = *(const v8h*)(bBase0 + k), b0h = *(const v8h*)(bBase0 + k + 16);
    v8h b1l = *(const v8h*)(bBase1 + k), b1h = *(const v8h*)(bBase1 + k + 16);
    v8h b2l = *(const v8h*)(bBase2 + k), b2h = *(const v8h*)(bBase2 + k + 16);
    v8h b3l = *(const v8h*)(bBase3 + k), b3h = *(const v8h*)(bBase3 + k + 16);
    v16h b0 = __builtin_shufflevector(b0l, b0h, 0,1,2,3,4,5,6,7,8,9,10,11,12,13,14,15);
    v16h b1 = __builtin_shufflevector(b1l, b1h, 0,1,2,3,4,5,6,7,8,9,10,11,12,13,14,15);
    v16h b2 = __builtin_shufflevector(b2l, b2h, 0,1,2,3,4,5,6,7,8,9,10,11,12,13,14,15);
    v16h b3 = __builtin_shufflevector(b3l, b3h, 0,1,2,3,4,5,6,7,8,9,10,11,12,13,14,15);

    acc[0] = __builtin_amdgcn_wmma_f32_16x16x32_f16(false, a, false, b0, (short)0, acc[0], false, false);
    acc[1] = __builtin_amdgcn_wmma_f32_16x16x32_f16(false, a, false, b1, (short)0, acc[1], false, false);
    acc[2] = __builtin_amdgcn_wmma_f32_16x16x32_f16(false, a, false, b2, (short)0, acc[2], false, false);
    acc[3] = __builtin_amdgcn_wmma_f32_16x16x32_f16(false, a, false, b3, (short)0, acc[3], false, false);
  }

  // C/D layout: VGPR i -> row = tileM*16 + i + 8*sel, col = grp*64 + j*16 + mn
#pragma unroll
  for (int j = 0; j < 4; ++j) {
    int col = grp * 64 + j * 16 + mn;
    if (col < WNC) {
      float bv = bias[col];
      int rbase = tileM * 16 + sel * 8;
#pragma unroll
      for (int i = 0; i < 8; ++i)
        out[(size_t)(rbase + i) * WNC + col] = acc[j][i] + bv;
    }
  }
}

// ---------------------------------------------------------------------------
extern "C" void kernel_launch(void* const* d_in, const int* in_sizes, int n_in,
                              void* d_out, int out_size, void* d_ws, size_t ws_size,
                              hipStream_t stream) {
  const float* x     = (const float*)d_in[0];
  const float* t0    = (const float*)d_in[1];
  const float* sc0   = (const float*)d_in[2];
  const float* w0    = (const float*)d_in[3];
  const float* g0    = (const float*)d_in[4];
  const float* b0    = (const float*)d_in[5];
  const float* t1    = (const float*)d_in[6];
  const float* sc1   = (const float*)d_in[7];
  const float* w1    = (const float*)d_in[8];
  const float* g1    = (const float*)d_in[9];
  const float* b1    = (const float*)d_in[10];
  const float* t2    = (const float*)d_in[11];
  const float* sc2   = (const float*)d_in[12];
  const float* w2    = (const float*)d_in[13];
  const float* g2    = (const float*)d_in[14];
  const float* b2    = (const float*)d_in[15];
  const float* cls_w = (const float*)d_in[16];
  const float* cls_b = (const float*)d_in[17];
  float* out = (float*)d_out;

  // scratch: two 1MB f32 ping-pong buffers; f16 views reuse them once the
  // f32 producers are consumed (stream ordering guarantees safety).
  float* bufA = (float*)d_ws;                   // 512*512 f32 (1 MB)
  float* bufB = bufA + (size_t)WB * WHID;       // 512*512 f32 (1 MB)
  _Float16* fh = (_Float16*)bufB;               // 512*512 f16 after LN3
  _Float16* wh = (_Float16*)bufA;               // 1024*512 f16 after wav3

  dim3 wgrid(WHID / OT, WB / BT);               // (32, 8)

  // layer 0
  wavelet_kernel<<<wgrid, 256, 0, stream>>>(x, t0, sc0, w0, bufA, WIN0);
  silu_ln_kernel<float><<<WB, 256, 0, stream>>>(bufA, g0, b0, bufB);
  // layer 1
  wavelet_kernel<<<wgrid, 256, 0, stream>>>(bufB, t1, sc1, w1, bufA, WHID);
  silu_ln_kernel<float><<<WB, 256, 0, stream>>>(bufA, g1, b1, bufB);
  // layer 2
  wavelet_kernel<<<wgrid, 256, 0, stream>>>(bufB, t2, sc2, w2, bufA, WHID);
  silu_ln_kernel<_Float16><<<WB, 256, 0, stream>>>(bufA, g2, b2, fh);
  // classifier: convert weights (overwrites bufA, no longer needed), then WMMA
  convert_w_f16<<<(WNCP * WHID) / 256, 256, 0, stream>>>(cls_w, wh);
  cls_gemm_wmma<<<(32 * (WNCP / 64)) / 8, 256, 0, stream>>>(fh, wh, cls_b, out);
}
